// IIBNet_89240830476536
// MI455X (gfx1250) — matrix operand order repaired and checked by reading, not compiled, not run on hardware
//
#include <hip/hip_runtime.h>
#include <math.h>

// ---------------------------------------------------------------------------
// Problem constants (from reference): B=4, C=128, H=W=224, NUM_CLASSES=41
// ---------------------------------------------------------------------------
#define BB 4
#define CC 128
#define HH 224
#define WW 224
#define HW (HH * WW)            // 50176
#define NCLS 41
#define CIN2 256                // concat(add, mul) channels
#define CP 128                  // channel pairs of CIN2 (uint = 2 bf16)
#define TILES_X (WW / 32)       // 7
#define TILES_Y (HH / 2)        // 112 (2 output rows per block)

typedef __attribute__((ext_vector_type(16))) __bf16 v16bf;
typedef __attribute__((ext_vector_type(8)))  float  v8f;

union FragBF { uint4 q[2]; unsigned int u[8]; v16bf v; };

__device__ __forceinline__ unsigned short f2bf(float f) {
    unsigned int u = __float_as_uint(f);
    unsigned int r = u + 0x7FFFu + ((u >> 16) & 1u);   // round-to-nearest-even
    return (unsigned short)(r >> 16);
}

// async global->LDS copies (CDNA5, ASYNCcnt); inline asm because the clang
// builtin arity differs across toolchains (see bridge doc).
__device__ __forceinline__ void async_ld_b128(unsigned int ldsAddr,
                                              const void* gaddr) {
    asm volatile("global_load_async_to_lds_b128 %0, %1, off"
                 :: "v"(ldsAddr), "v"((unsigned long long)gaddr) : "memory");
}
__device__ __forceinline__ void async_ld_b32(unsigned int ldsAddr,
                                             const void* gaddr) {
    asm volatile("global_load_async_to_lds_b32 %0, %1, off"
                 :: "v"(ldsAddr), "v"((unsigned long long)gaddr) : "memory");
}
__device__ __forceinline__ void wait_async0() {
    asm volatile("s_wait_asynccnt 0x0" ::: "memory");
}

// ---------------------------------------------------------------------------
// 1) pack X: uint[b][cp][y][x], cp<64 -> add channels (2cp,2cp+1),
//            cp>=64 -> mul channels (2(cp-64), +1). bf16 pair, even K in [15:0].
// ---------------------------------------------------------------------------
__global__ __launch_bounds__(256) void pack_x_kernel(
    const float* __restrict__ r, const float* __restrict__ d,
    unsigned int* __restrict__ Xp)
{
    int i = blockIdx.x * 256 + threadIdx.x;
    const int TOT = BB * CP * HW;
    if (i >= TOT) return;
    int x  = i % WW;
    int y  = (i / WW) % HH;
    int cp = (i / HW) % CP;
    int b  = i / (CP * HW);
    int  c;
    bool isMul;
    if (cp < 64) { c = cp * 2;        isMul = false; }
    else         { c = (cp - 64) * 2; isMul = true;  }
    int idx = ((b * CC + c) * HH + y) * WW + x;
    float r0 = r[idx],      r1 = r[idx + HW];
    float d0 = d[idx],      d1 = d[idx + HW];
    float v0, v1;
    if (!isMul) { v0 = r0 + d0;                     v1 = r1 + d1; }
    else        { v0 = r0 * (1.0f / (1.0f + expf(-d0)));
                  v1 = r1 * (1.0f / (1.0f + expf(-d1))); }
    unsigned int lo = f2bf(v0), hi = f2bf(v1);
    Xp[i] = lo | (hi << 16);
}

// ---------------------------------------------------------------------------
// 2) pack W: uint[t=ky*3+kx][o][cp]  (bf16 pair over adjacent cin)
//    w_fuse layout: [128][256][3][3] f32
// ---------------------------------------------------------------------------
__global__ __launch_bounds__(256) void pack_w_kernel(
    const float* __restrict__ w, unsigned int* __restrict__ Wp)
{
    int i = blockIdx.x * 256 + threadIdx.x;
    const int TOT = 9 * CC * CP;    // 147456
    if (i >= TOT) return;
    int cp = i % CP;
    int o  = (i / CP) % CC;
    int t  = i / (CP * CC);
    int ky = t / 3, kx = t % 3;
    int ci = cp * 2;
    float w0 = w[((o * CIN2 + ci    ) * 3 + ky) * 3 + kx];
    float w1 = w[((o * CIN2 + ci + 1) * 3 + ky) * 3 + kx];
    Wp[i] = (unsigned int)f2bf(w0) | ((unsigned int)f2bf(w1) << 16);
}

// ---------------------------------------------------------------------------
// 3) conv via implicit GEMM on v_wmma_f32_16x16x32_bf16.
//    Block = 128 thr = 4 waves; tile = 128 Cout x (2 rows x 32 cols).
//    Wave w: Cout [32w, 32w+32) x all 64 px -> 8 accumulators, 8 WMMA / tap.
//    X chunk (4 halo rows x 34 cols x 32 cin) staged once per cin-chunk via
//    async b32 loads; weights staged per tap via async b128 loads. Both land
//    in LDS directly (ASYNCcnt), no VGPR round trip.
// ---------------------------------------------------------------------------
__global__ __launch_bounds__(128) void conv_wmma_kernel(
    const unsigned int* __restrict__ Xp,  // [B][CP][H][W] (bf16 cin pairs)
    const unsigned int* __restrict__ Wp,  // [9][128][128]
    float* __restrict__ fuse)             // [B][128][H][W]
{
    __shared__ __align__(16) unsigned int wA[128][20];     // 16 data + 4 pad
    __shared__ __align__(16) unsigned int xH[4][34][20];   // rows, cols(+halo)

    int blk = blockIdx.x;
    int x0  = (blk % TILES_X) * 32;
    int y0  = ((blk / TILES_X) % TILES_Y) * 2;
    int b   = blk / (TILES_X * TILES_Y);

    int tid  = threadIdx.x;
    int lane = tid & 31;
    int wave = tid >> 5;          // 0..3 -> Cout base = wave*32
    int m    = lane & 15;
    int hi   = lane >> 4;         // lane-half select

    v8f acc[2][2][2] = {};        // [row][colgrp][mfrag]

    for (int kc = 0; kc < 8; ++kc) {          // cin chunk of 32 (16 uint pairs)
        __syncthreads();                      // prior reads of xH/wA done
        // stage X halo tile: 4 rows x 34 cols x 16 uints = 2176 (17 per thread)
        for (int i = tid; i < 4 * 34 * 16; i += 128) {
            int col = i % 34;
            int t   = i / 34;
            int j   = t & 15;
            int row = t >> 4;
            int yy  = y0 + row - 1;
            int xx  = x0 + col - 1;
            if (yy >= 0 && yy < HH && xx >= 0 && xx < WW) {
                async_ld_b32((unsigned int)(size_t)&xH[row][col][j],
                             Xp + ((b * CP + kc * 16 + j) * HH + yy) * WW + xx);
            } else {
                xH[row][col][j] = 0u;         // zero halo
            }
        }
        for (int t9 = 0; t9 < 9; ++t9) {      // filter taps
            if (t9) __syncthreads();          // prior tap's wA reads done
            const unsigned int* wsrc = Wp + t9 * (CC * CP) + kc * 16;
            // 512 x b128 chunks: o = q>>2, 16B sub-chunk = q&3 (4 per thread)
            for (int q = tid; q < 512; q += 128) {
                int o = q >> 2, sub = q & 3;
                async_ld_b128((unsigned int)(size_t)&wA[o][sub * 4],
                              wsrc + o * CP + sub * 4);
            }
            wait_async0();                    // this wave's async copies done
            __syncthreads();                  // all waves' staging visible

            // A fragments (16x32 bf16): uint idx = (v>=4?8:0)+(hi?4:0)+(v&3)
            FragBF A[2];
            #pragma unroll
            for (int mf = 0; mf < 2; ++mf) {
                const unsigned int* wrow = &wA[wave * 32 + mf * 16 + m][0];
                A[mf].q[0] = *(const uint4*)(wrow + hi * 4);
                A[mf].q[1] = *(const uint4*)(wrow + 8 + hi * 4);
            }
            int ky = t9 / 3, kx = t9 % 3;
            // B fragments (32x16 bf16): uint idx = (v>=4?8:0)+(hi?4:0)+(v&3)
            #pragma unroll
            for (int yr = 0; yr < 2; ++yr) {
                #pragma unroll
                for (int g = 0; g < 2; ++g) {
                    FragBF Bf;
                    const unsigned int* xcol = &xH[yr + ky][g * 16 + m + kx][0];
                    Bf.q[0] = *(const uint4*)(xcol + hi * 4);
                    Bf.q[1] = *(const uint4*)(xcol + 8 + hi * 4);
                    #pragma unroll
                    for (int mf = 0; mf < 2; ++mf)
                        acc[yr][g][mf] = __builtin_amdgcn_wmma_f32_16x16x32_bf16(
                            false, A[mf].v, false, Bf.v, (short)0,
                            acc[yr][g][mf], false, false);
                }
            }
        }
    }

    // store: D 16x16 f32 -> row = v + hi*8, col = m
    #pragma unroll
    for (int yr = 0; yr < 2; ++yr)
        #pragma unroll
        for (int g = 0; g < 2; ++g)
            #pragma unroll
            for (int mf = 0; mf < 2; ++mf)
                #pragma unroll
                for (int v = 0; v < 8; ++v) {
                    int o = wave * 32 + mf * 16 + v + hi * 8;
                    fuse[((size_t)(b * CC + o) * HH + (y0 + yr)) * WW
                         + x0 + g * 16 + m] = acc[yr][g][mf][v];
                }
}

// ---------------------------------------------------------------------------
// 4) zero-init stats workspace
// ---------------------------------------------------------------------------
__global__ __launch_bounds__(256) void zero_kernel(float* p, int n) {
    int i = blockIdx.x * 256 + threadIdx.x;
    if (i < n) p[i] = 0.0f;
}

// ---------------------------------------------------------------------------
// 5) segment stats: bins[b][41][128] += feat, cnt[b][41] += 1 per pixel
//    labels_ds[b][y][x] = label[b][4y][4x]  (label is [4][896][896] int32)
// ---------------------------------------------------------------------------
#define SROWS 8
__global__ __launch_bounds__(256) void stats_kernel(
    const float* __restrict__ feat, const int* __restrict__ label,
    float* __restrict__ bins, float* __restrict__ cnt)
{
    __shared__ float sb[NCLS * CC];
    __shared__ float sc[NCLS];
    __shared__ unsigned char slab[SROWS * WW];

    int tiles = HH / SROWS;                 // 28
    int b  = blockIdx.x / tiles;
    int y0 = (blockIdx.x % tiles) * SROWS;
    int tid = threadIdx.x;

    for (int i = tid; i < NCLS * CC; i += 256) sb[i] = 0.0f;
    for (int i = tid; i < NCLS; i += 256) sc[i] = 0.0f;
    for (int p = tid; p < SROWS * WW; p += 256) {
        int y = y0 + p / WW, x = p % WW;
        slab[p] = (unsigned char)label[((size_t)b * 896 + 4 * y) * 896 + 4 * x];
    }
    __syncthreads();

    for (int p = tid; p < SROWS * WW; p += 256)
        atomicAdd(&sc[slab[p]], 1.0f);

    for (int c = 0; c < CC; ++c) {
        const float* src = feat + ((size_t)(b * CC + c) * HH + y0) * WW;
        for (int p = tid; p < SROWS * WW; p += 256)
            atomicAdd(&sb[(int)slab[p] * CC + c], src[p]);
    }
    __syncthreads();

    for (int i = tid; i < NCLS * CC; i += 256)
        atomicAdd(&bins[(size_t)b * NCLS * CC + i], sb[i]);
    for (int i = tid; i < NCLS; i += 256)
        atomicAdd(&cnt[b * NCLS + i], sc[i]);
}

// ---------------------------------------------------------------------------
// 6) finalize RA: class-norm -> att -> relu(att@w1^T) -> sigmoid(hid@w2^T)
//    (empty segments: probability ~0 with 50176 uniform pixels over 41 classes;
//     emit 0 there — the reference value is unreproducible PRNG noise anyway)
// ---------------------------------------------------------------------------
__global__ __launch_bounds__(128) void finalize_kernel(
    const float* __restrict__ bins, const float* __restrict__ cnt,
    const float* __restrict__ w1, const float* __restrict__ w2,
    float* __restrict__ s)
{
    __shared__ float att_s[CC];
    __shared__ float hid_s[8];
    int b = blockIdx.x, c = threadIdx.x;

    float sum = 0.0f, sumsq = 0.0f;
    for (int l = 0; l < NCLS; ++l) {
        float ct = cnt[b * NCLS + l];
        float q  = (ct > 0.0f) ? bins[((size_t)b * NCLS + l) * CC + c] / ct : 0.0f;
        sum   += q;
        sumsq += q * q;
    }
    att_s[c] = sum / fmaxf(sqrtf(sumsq), 1e-12f);
    __syncthreads();
    if (c < 8) {
        float h = 0.0f;
        for (int j = 0; j < CC; ++j) h += att_s[j] * w1[c * CC + j];
        hid_s[c] = fmaxf(h, 0.0f);
    }
    __syncthreads();
    float o = 0.0f;
    for (int j = 0; j < 8; ++j) o += hid_s[j] * w2[c * 8 + j];
    s[b * CC + c] = 1.0f / (1.0f + expf(-o));
}

// ---------------------------------------------------------------------------
// 7) combine: out = fuse * s_f[b,c] + d * s_d[b,c]   (float4 vectorized)
// ---------------------------------------------------------------------------
__global__ __launch_bounds__(256) void combine_kernel(
    const float* __restrict__ fuse, const float* __restrict__ d,
    const float* __restrict__ s_f, const float* __restrict__ s_d,
    float* __restrict__ out)
{
    const int N4 = BB * CC * HW / 4;
    int i = blockIdx.x * 256 + threadIdx.x;
    if (i >= N4) return;
    int e = i * 4;
    int c = (e / HW) % CC;
    int b = e / (CC * HW);
    float sf = s_f[b * CC + c];
    float sd = s_d[b * CC + c];
    const float4 fv = ((const float4*)fuse)[i];
    const float4 dv = ((const float4*)d)[i];
    float4 ov;
    ov.x = fv.x * sf + dv.x * sd;
    ov.y = fv.y * sf + dv.y * sd;
    ov.z = fv.z * sf + dv.z * sd;
    ov.w = fv.w * sf + dv.w * sd;
    ((float4*)out)[i] = ov;
}

// ---------------------------------------------------------------------------
extern "C" void kernel_launch(void* const* d_in, const int* in_sizes, int n_in,
                              void* d_out, int out_size, void* d_ws, size_t ws_size,
                              hipStream_t stream)
{
    const float* r      = (const float*)d_in[0];
    const float* d      = (const float*)d_in[1];
    const int*   label  = (const int*)  d_in[2];
    const float* w_fuse = (const float*)d_in[3];
    const float* w1_d   = (const float*)d_in[4];
    const float* w2_d   = (const float*)d_in[5];
    const float* w1_f   = (const float*)d_in[6];
    const float* w2_f   = (const float*)d_in[7];
    float* out = (float*)d_out;

    auto al = [](size_t v) { return (v + 255) & ~(size_t)255; };
    char* base = (char*)d_ws;
    size_t off = 0;
    unsigned int* Xp   = (unsigned int*)(base + off); off = al(off + (size_t)BB * CP * HW * 4);
    unsigned int* Wp   = (unsigned int*)(base + off); off = al(off + (size_t)9 * CC * CP * 4);
    float*        fuse = (float*)(base + off);        off = al(off + (size_t)BB * CC * HW * 4);
    float*        bins_d = (float*)(base + off);      off = al(off + (size_t)BB * NCLS * CC * 4);
    float*        cnt_d  = (float*)(base + off);      off = al(off + (size_t)BB * NCLS * 4);
    float*        bins_f = (float*)(base + off);      off = al(off + (size_t)BB * NCLS * CC * 4);
    float*        cnt_f  = (float*)(base + off);      off = al(off + (size_t)BB * NCLS * 4);
    float*        s_d    = (float*)(base + off);      off = al(off + (size_t)BB * CC * 4);
    float*        s_f    = (float*)(base + off);      off = al(off + (size_t)BB * CC * 4);
    (void)ws_size; (void)in_sizes; (void)n_in; (void)out_size;

    // stats area is contiguous from bins_d; zero it all
    int statsN = (int)(((char*)s_f + al((size_t)BB * CC * 4)) - (char*)bins_d) / 4;

    {   // pack X
        int tot = BB * CP * HW;
        pack_x_kernel<<<(tot + 255) / 256, 256, 0, stream>>>(r, d, Xp);
    }
    {   // pack W
        int tot = 9 * CC * CP;
        pack_w_kernel<<<(tot + 255) / 256, 256, 0, stream>>>(w_fuse, Wp);
    }
    zero_kernel<<<(statsN + 255) / 256, 256, 0, stream>>>(bins_d, statsN);

    conv_wmma_kernel<<<BB * TILES_Y * TILES_X, 128, 0, stream>>>(Xp, Wp, fuse);

    stats_kernel<<<BB * (HH / SROWS), 256, 0, stream>>>(d,    label, bins_d, cnt_d);
    stats_kernel<<<BB * (HH / SROWS), 256, 0, stream>>>(fuse, label, bins_f, cnt_f);

    finalize_kernel<<<BB, 128, 0, stream>>>(bins_d, cnt_d, w1_d, w2_d, s_d);
    finalize_kernel<<<BB, 128, 0, stream>>>(bins_f, cnt_f, w1_f, w2_f, s_f);

    {
        int n4 = BB * CC * HW / 4;
        combine_kernel<<<(n4 + 255) / 256, 256, 0, stream>>>(fuse, d, s_f, s_d, out);
    }
}
